// CDTree_19215683682931
// MI455X (gfx1250) — matrix operand order repaired and checked by reading, not compiled, not run on hardware
//
#include <hip/hip_runtime.h>

// ---------------------------------------------------------------------------
// CDNA5 (gfx1250) soft-decision-tree forward.
// Convs = implicit GEMM on v_wmma_f32_16x16x32_f16, templated per depth so all
// index math is constant (no runtime int division). Weights are pre-swizzled
// into per-lane WMMA fragment order so B operands load as single v16h (b128x2)
// loads. Deep nodes keep NT accumulators per wave to reuse the im2col A
// fragment across NT WMMAs.
// ---------------------------------------------------------------------------

typedef __attribute__((ext_vector_type(16))) _Float16 v16h;
typedef __attribute__((ext_vector_type(8)))  float    v8f;
typedef __attribute__((ext_vector_type(4)))  float    v4f;

#define BATCH 256
#define OUTC  10

struct PtrArr16 { const float* p[16]; };

// ---------------------------------------------------------------------------
// Weight prep: f32 [OC][K] -> f16 fragments [kc][nt][lane][e] matching the
// contiguous-k B order this file uses: lane -> col N = nt*16 + (lane&15); elem e -> K = kc*32 +
// 16*(lane>>4) + e. Zero padded in K and N.
// ---------------------------------------------------------------------------
__global__ void prep_w_swz(const float* __restrict__ w, _Float16* __restrict__ wB,
                           int OC, int K, int NT, int total) {
  int idx = blockIdx.x * blockDim.x + threadIdx.x;
  if (idx >= total) return;
  int e    = idx & 15;
  int lane = (idx >> 4) & 31;
  int t    = idx >> 9;          // kc*NT + nt
  int nt   = t % NT;
  int kc   = t / NT;
  int k = kc * 32 + (lane >> 4) * 16 + e;
  int n = nt * 16 + (lane & 15);
  float v = (k < K && n < OC) ? w[n * K + k] : 0.0f;
  wB[idx] = (_Float16)v;
}

// ---------------------------------------------------------------------------
// Implicit-GEMM conv3x3 VALID, NCHW input (square H), f32 out [M][OC] + bias.
// One wave per 16-row M tile; NT=ocPad/16 column tiles accumulated per wave,
// reusing one im2col A fragment per K chunk.
// A-frag (16x32 f16): lane -> row M=lane&15; hi=lane>>4; e -> K = 16*hi + e
// C/D  (16x16 f32):   lane -> col N=lane&15; vgpr r -> row M = r + 8*hi
// ---------------------------------------------------------------------------
template <int C, int H, int OH, int OC, int ocPad, int K, int Kpad, int NT>
__global__ void conv_wmma_kernel(const float* __restrict__ in, const _Float16* __restrict__ wB,
                                 const float* __restrict__ bias, float* __restrict__ out,
                                 int Mtiles) {
  constexpr int OW  = OH;
  constexpr int M   = BATCH * OH * OW;
  constexpr int KCH = Kpad / 32;
  const int lane  = threadIdx.x & 31;
  const int wave  = threadIdx.x >> 5;
  const int mtile = blockIdx.x * 4 + wave;
  if (mtile >= Mtiles) return;
  const int row = lane & 15;
  const int hi  = lane >> 4;
  const int m   = mtile * 16 + row;
  const bool mvalid = (m < M);

  int b = 0, oy = 0, ox = 0;
  if (mvalid) { ox = m % OW; int t = m / OW; oy = t % OH; b = t / OH; }
  const float* inb = in + ((size_t)b * C) * (H * H) + oy * H + ox;

  v8f acc[NT];
#pragma unroll
  for (int nt = 0; nt < NT; ++nt) acc[nt] = v8f{};

  const v16h* wfrag = (const v16h*)wB;
  for (int kc = 0; kc < KCH; ++kc) {
    v16h a;
#pragma unroll
    for (int e = 0; e < 16; ++e) {
      int k = kc * 32 + hi * 16 + e;
      float v = 0.0f;
      if (mvalid && k < K) {
        int ic = k / 9, r = k - ic * 9;     // constant divisors -> mul/shift
        int ky = r / 3, kx = r - 3 * ky;
        v = inb[ic * (H * H) + ky * H + kx];
      }
      a[e] = (_Float16)v;
    }
#pragma unroll
    for (int nt = 0; nt < NT; ++nt) {
      v16h bf = wfrag[(kc * NT + nt) * 32 + lane];
      acc[nt] = __builtin_amdgcn_wmma_f32_16x16x32_f16(false, a, false, bf, (short)0,
                                                       acc[nt], false, false);
    }
  }

#pragma unroll
  for (int nt = 0; nt < NT; ++nt) {
    const int n = nt * 16 + row;
#pragma unroll
    for (int r = 0; r < 8; ++r) {
      int mm = mtile * 16 + hi * 8 + r;
      if (mm < M && n < OC) out[(size_t)mm * OC + n] = acc[nt][r] + bias[n];
    }
  }
}

// ---------------------------------------------------------------------------
// Per-channel batch-norm statistics over [M][OC] conv output.
// ---------------------------------------------------------------------------
__global__ void bn_stats_kernel(const float* __restrict__ conv, int M, int OC,
                                float* __restrict__ mean, float* __restrict__ rinv) {
  const int c = blockIdx.x;
  float s = 0.0f, sq = 0.0f;
  for (int i = threadIdx.x; i < M; i += blockDim.x) {
    float v = conv[(size_t)i * OC + c];
    s += v; sq += v * v;
  }
  __shared__ float ls[256], lq[256];
  ls[threadIdx.x] = s; lq[threadIdx.x] = sq;
  __syncthreads();
  for (int st = 128; st > 0; st >>= 1) {
    if (threadIdx.x < (unsigned)st) { ls[threadIdx.x] += ls[threadIdx.x + st];
                                      lq[threadIdx.x] += lq[threadIdx.x + st]; }
    __syncthreads();
  }
  if (threadIdx.x == 0) {
    float mu  = ls[0] / (float)M;
    float var = lq[0] / (float)M - mu * mu;
    mean[c] = mu;
    rinv[c] = rsqrtf(var + 1e-5f);
  }
}

// ---------------------------------------------------------------------------
// Normalize + affine + ReLU + 2x2/2 avgpool; output NCHW f32.
// ---------------------------------------------------------------------------
__global__ void bn_pool_kernel(const float* __restrict__ conv, const float* __restrict__ mean,
                               const float* __restrict__ rinv, const float* __restrict__ g,
                               const float* __restrict__ bb, float* __restrict__ pooled,
                               int OC, int CH, int PH, int total) {
  int idx = blockIdx.x * blockDim.x + threadIdx.x;
  if (idx >= total) return;
  int px = idx % PH; int t = idx / PH;
  int py = t % PH;   t /= PH;
  int c  = t % OC;   int b = t / OC;
  float mu = mean[c], ri = rinv[c], gg = g[c], be = bb[c];
  float s = 0.0f;
#pragma unroll
  for (int dy = 0; dy < 2; ++dy)
#pragma unroll
    for (int dx = 0; dx < 2; ++dx) {
      int cy = 2 * py + dy, cx = 2 * px + dx;
      float v = conv[((size_t)(b * CH + cy) * CH + cx) * OC + c];
      v = (v - mu) * ri * gg + be;
      s += fmaxf(v, 0.0f);
    }
  pooled[((size_t)(b * OC + c) * PH + py) * PH + px] = 0.25f * s;
}

// ---------------------------------------------------------------------------
// Gate FC: prob[b] = sigmoid(beta * (h[b,:] . w + fcb)).  grid = BATCH blocks.
// ---------------------------------------------------------------------------
__global__ void fc_sigmoid_kernel(const float* __restrict__ h, const float* __restrict__ w,
                                  const float* __restrict__ fcb, const float* __restrict__ beta,
                                  float* __restrict__ prob, int F) {
  int b = blockIdx.x;
  float s = 0.0f;
  for (int k = threadIdx.x; k < F; k += blockDim.x) s += h[(size_t)b * F + k] * w[k];
  __shared__ float ls[256];
  ls[threadIdx.x] = s;
  __syncthreads();
  for (int st = 128; st > 0; st >>= 1) {
    if (threadIdx.x < (unsigned)st) ls[threadIdx.x] += ls[threadIdx.x + st];
    __syncthreads();
  }
  if (threadIdx.x == 0) {
    float o = (ls[0] + fcb[0]) * beta[0];
    prob[b] = 1.0f / (1.0f + expf(-o));
  }
}

// ---------------------------------------------------------------------------
// Heap path probabilities: ppAll[(i-1)*B + b] for nodes i = 1..31.
// ---------------------------------------------------------------------------
__global__ void pathprob_kernel(const float* __restrict__ probAll, float* __restrict__ ppAll) {
  int b = threadIdx.x;
  float pp[32];
  pp[1] = 0.5f;
  for (int i = 1; i < 16; ++i) {
    float p = probAll[(i - 1) * BATCH + b];
    pp[2 * i]     = pp[i] * (1.0f - p);
    pp[2 * i + 1] = pp[i] * p;
  }
  for (int i = 1; i < 32; ++i) ppAll[(i - 1) * BATCH + b] = pp[i];
}

// ---------------------------------------------------------------------------
// Penalties for nodes 1..7: pen = sum(prob*pp)/sum(pp).
// ---------------------------------------------------------------------------
__global__ void pens_kernel(const float* __restrict__ probAll, const float* __restrict__ ppAll,
                            float* __restrict__ pen) {
  int b = threadIdx.x;
  __shared__ float sn[256], sd[256];
  for (int i = 1; i <= 7; ++i) {
    float p = probAll[(i - 1) * BATCH + b];
    float q = ppAll[(i - 1) * BATCH + b];
    sn[b] = p * q; sd[b] = q;
    __syncthreads();
    for (int st = 128; st > 0; st >>= 1) {
      if (b < st) { sn[b] += sn[b + st]; sd[b] += sd[b + st]; }
      __syncthreads();
    }
    if (b == 0) pen[i - 1] = sn[0] / sd[0];
    __syncthreads();
  }
}

// ---------------------------------------------------------------------------
// Leaf linears via WMMA: Q[leaf][b][10] = h_parent[b,:256] @ W^T + bias.
// h4 = depth-4 pooled buffers, contiguous [8][BATCH][256]. Leaf weights are
// pre-swizzled fragments [leaf][kc][lane][e]; A loads are 4x aligned float4.
// ---------------------------------------------------------------------------
__global__ void leaf_wmma_kernel(const float* __restrict__ h4, const _Float16* __restrict__ wB,
                                 PtrArr16 lb, float* __restrict__ Q) {
  const int lane  = threadIdx.x & 31;
  const int wave  = threadIdx.x >> 5;
  const int mtile = blockIdx.x * 4 + wave;   // 0..15
  const int leaf  = blockIdx.y;              // 0..15
  const float* h      = h4 + (size_t)(leaf >> 1) * (BATCH * 256);
  const v16h*  wfrag  = (const v16h*)wB + (size_t)leaf * (8 * 32);
  const int row = lane & 15, hi = lane >> 4;
  const int m   = mtile * 16 + row;

  v8f acc = {};
#pragma unroll
  for (int kc = 0; kc < 8; ++kc) {
    const float* hp = h + (size_t)m * 256 + kc * 32 + hi * 16;
    v4f f0 = *(const v4f*)(hp);
    v4f f1 = *(const v4f*)(hp + 4);
    v4f f2 = *(const v4f*)(hp + 8);
    v4f f3 = *(const v4f*)(hp + 12);
    v16h a;
#pragma unroll
    for (int e = 0; e < 4; ++e) {
      a[e]      = (_Float16)f0[e];
      a[e + 4]  = (_Float16)f1[e];
      a[e + 8]  = (_Float16)f2[e];
      a[e + 12] = (_Float16)f3[e];
    }
    v16h bf = wfrag[kc * 32 + lane];
    acc = __builtin_amdgcn_wmma_f32_16x16x32_f16(false, a, false, bf, (short)0, acc,
                                                 false, false);
  }
  const float* bias = lb.p[leaf];
  if (row < OUTC) {
#pragma unroll
    for (int r = 0; r < 8; ++r) {
      int mm = mtile * 16 + hi * 8 + r;
      Q[((size_t)leaf * BATCH + mm) * OUTC + row] = acc[r] + bias[row];
    }
  }
}

// ---------------------------------------------------------------------------
// Per-leaf cross-entropy mean and mean path prob.
// ---------------------------------------------------------------------------
__global__ void leaf_stats_kernel(const float* __restrict__ Q, const int* __restrict__ y,
                                  const float* __restrict__ ppAll, float* __restrict__ ce,
                                  float* __restrict__ mp) {
  int leaf = blockIdx.x;
  int b = threadIdx.x;
  const float* q = Q + ((size_t)leaf * BATCH + b) * OUTC;
  float mx = q[0];
  for (int o = 1; o < OUTC; ++o) mx = fmaxf(mx, q[o]);
  float se = 0.0f;
  for (int o = 0; o < OUTC; ++o) se += expf(q[o] - mx);
  float lse = mx + logf(se);
  int yy = y[b];
  float lp = q[yy] - lse;
  float pp = ppAll[(15 + leaf) * BATCH + b];
  __shared__ float s1[256], s2[256];
  s1[b] = lp; s2[b] = pp;
  __syncthreads();
  for (int st = 128; st > 0; st >>= 1) {
    if (b < st) { s1[b] += s1[b + st]; s2[b] += s2[b + st]; }
    __syncthreads();
  }
  if (b == 0) { ce[leaf] = -s1[0] / (float)BATCH; mp[leaf] = s2[0] / (float)BATCH; }
}

// ---------------------------------------------------------------------------
// Output gather (argmax leaf per sample) + scalar total.
// ---------------------------------------------------------------------------
__global__ void final_kernel(const float* __restrict__ Q, const float* __restrict__ ppAll,
                             const float* __restrict__ ce, const float* __restrict__ mp,
                             const float* __restrict__ pen, float* __restrict__ out) {
  int b = threadIdx.x;
  float best = -3.4e38f;
  int idx = 0;
  for (int l = 0; l < 16; ++l) {
    float v = ppAll[(15 + l) * BATCH + b];
    if (v != v) v = 0.0f;               // nan_to_num
    if (v > best) { best = v; idx = l; }
  }
  for (int o = 0; o < OUTC; ++o)
    out[b * OUTC + o] = Q[((size_t)idx * BATCH + b) * OUTC + o];
  if (b == 0) {
    float total = 0.001f;
    for (int l = 0; l < 16; ++l) total += ce[l] * mp[l];
    for (int i = 1; i <= 7; ++i) {
      int d = (i >= 4) ? 3 : ((i >= 2) ? 2 : 1);
      float lmb = 0.1f * exp2f(-(float)d);
      float p = pen[i - 1];
      total -= lmb * 0.5f * (logf(p) + logf(1.0f - p));
    }
    out[BATCH * OUTC] = total;
  }
}

// ---------------------------------------------------------------------------
// Host side
// ---------------------------------------------------------------------------
struct NodeDim { int C, H, OC, OH, PH, K, Kpad, ocPad, NT; };
static const NodeDim DN[5] = {
  {0, 0, 0, 0, 0, 0, 0, 0, 0},
  {3, 64, 8, 62, 31, 27, 32, 16, 1},     // depth 1
  {8, 31, 16, 29, 14, 72, 96, 16, 1},    // depth 2
  {16, 14, 32, 12, 6, 144, 160, 32, 2},  // depth 3
  {32, 6, 64, 4, 2, 288, 288, 64, 4},    // depth 4
};

static inline int depth_of(int i) { int d = 0; while (i) { d++; i >>= 1; } return d; }

extern "C" void kernel_launch(void* const* d_in, const int* in_sizes, int n_in,
                              void* d_out, int out_size, void* d_ws, size_t ws_size,
                              hipStream_t stream) {
  (void)in_sizes; (void)n_in; (void)out_size; (void)ws_size;

  const float* x = (const float*)d_in[0];
  const int*   y = (const int*)d_in[1];
  auto P = [&](int idx) { return (const float*)d_in[idx]; };
  // param index bases: node i (1..15): 2 + 7*(i-1); leaf l (0..15): 107 + 2*l

  // ---- carve workspace ----
  char* base = (char*)d_ws;
  size_t off = 0;
  auto alloc = [&](size_t bytes) -> char* {
    off = (off + 255) & ~(size_t)255;
    char* p = base + off;
    off += bytes;
    return p;
  };

  // conv scratch: max M*OC = (256*62*62)*8 floats (node 1)
  float* convS = (float*)alloc((size_t)(BATCH * 62 * 62) * 8 * sizeof(float));
  // pooled buffers; depth-4 (nodes 8..15) contiguous for leaf kernel
  float* pooled[16];
  for (int i = 1; i <= 7; ++i) {
    const NodeDim& nd = DN[depth_of(i)];
    pooled[i] = (float*)alloc((size_t)BATCH * nd.OC * nd.PH * nd.PH * sizeof(float));
  }
  float* pooled4 = (float*)alloc((size_t)8 * BATCH * 256 * sizeof(float));
  for (int i = 8; i <= 15; ++i) pooled[i] = pooled4 + (size_t)(i - 8) * BATCH * 256;

  _Float16* wBn[16];
  for (int i = 1; i <= 15; ++i) {
    const NodeDim& nd = DN[depth_of(i)];
    wBn[i] = (_Float16*)alloc((size_t)nd.Kpad * nd.ocPad * sizeof(_Float16));
  }
  _Float16* wBleaf = (_Float16*)alloc((size_t)16 * 256 * 16 * sizeof(_Float16));
  float* mean    = (float*)alloc(64 * sizeof(float));
  float* rinv    = (float*)alloc(64 * sizeof(float));
  float* probAll = (float*)alloc(15 * BATCH * sizeof(float));
  float* ppAll   = (float*)alloc(31 * BATCH * sizeof(float));
  float* pen     = (float*)alloc(7 * sizeof(float));
  float* Q       = (float*)alloc((size_t)16 * BATCH * OUTC * sizeof(float));
  float* ce      = (float*)alloc(16 * sizeof(float));
  float* mp      = (float*)alloc(16 * sizeof(float));

  // ---- weight prep (swizzled fragment layout) ----
  for (int i = 1; i <= 15; ++i) {
    const NodeDim& nd = DN[depth_of(i)];
    int pb = 2 + 7 * (i - 1);
    int total = (nd.Kpad / 32) * nd.NT * 32 * 16;   // == Kpad*ocPad
    prep_w_swz<<<(total + 255) / 256, 256, 0, stream>>>(P(pb + 0), wBn[i],
                                                        nd.OC, nd.K, nd.NT, total);
  }
  for (int l = 0; l < 16; ++l) {
    int pb = 107 + 2 * l;
    int total = (256 / 32) * 1 * 32 * 16;           // 4096
    prep_w_swz<<<(total + 255) / 256, 256, 0, stream>>>(P(pb), wBleaf + (size_t)l * 256 * 16,
                                                        OUTC, 256, 1, total);
  }

  // ---- inner-node pipeline ----
  for (int i = 1; i <= 15; ++i) {
    const int d = depth_of(i);
    const NodeDim& nd = DN[d];
    int pb = 2 + 7 * (i - 1);
    const float* in = (i == 1) ? x : pooled[i >> 1];
    int M = BATCH * nd.OH * nd.OH;
    int Mtiles = (M + 15) / 16;
    dim3 grid((Mtiles + 3) / 4);
    switch (d) {
      case 1:
        conv_wmma_kernel<3, 64, 62, 8, 16, 27, 32, 1><<<grid, 128, 0, stream>>>(
            in, wBn[i], P(pb + 1), convS, Mtiles);
        break;
      case 2:
        conv_wmma_kernel<8, 31, 29, 16, 16, 72, 96, 1><<<grid, 128, 0, stream>>>(
            in, wBn[i], P(pb + 1), convS, Mtiles);
        break;
      case 3:
        conv_wmma_kernel<16, 14, 12, 32, 32, 144, 160, 2><<<grid, 128, 0, stream>>>(
            in, wBn[i], P(pb + 1), convS, Mtiles);
        break;
      default:
        conv_wmma_kernel<32, 6, 4, 64, 64, 288, 288, 4><<<grid, 128, 0, stream>>>(
            in, wBn[i], P(pb + 1), convS, Mtiles);
        break;
    }
    bn_stats_kernel<<<nd.OC, 256, 0, stream>>>(convS, M, nd.OC, mean, rinv);
    int ptotal = BATCH * nd.OC * nd.PH * nd.PH;
    bn_pool_kernel<<<(ptotal + 255) / 256, 256, 0, stream>>>(convS, mean, rinv,
                                                             P(pb + 2), P(pb + 3), pooled[i],
                                                             nd.OC, nd.OH, nd.PH, ptotal);
    int F = nd.OC * nd.PH * nd.PH;
    fc_sigmoid_kernel<<<BATCH, 256, 0, stream>>>(pooled[i], P(pb + 4), P(pb + 5), P(pb + 6),
                                                 probAll + (i - 1) * BATCH, F);
  }

  // ---- tree bookkeeping ----
  pathprob_kernel<<<1, 256, 0, stream>>>(probAll, ppAll);
  pens_kernel<<<1, 256, 0, stream>>>(probAll, ppAll, pen);

  // ---- leaves ----
  PtrArr16 lb;
  for (int l = 0; l < 16; ++l) lb.p[l] = P(107 + 2 * l + 1);
  leaf_wmma_kernel<<<dim3(4, 16), 128, 0, stream>>>(pooled4, wBleaf, lb, Q);
  leaf_stats_kernel<<<16, 256, 0, stream>>>(Q, y, ppAll, ce, mp);

  // ---- output + total ----
  final_kernel<<<1, 256, 0, stream>>>(Q, ppAll, ce, mp, pen, (float*)d_out);
}
